// Mix_Loss_single_28321014350293
// MI455X (gfx1250) — matrix operand order repaired and checked by reading, not compiled, not run on hardware
//
#include <hip/hip_runtime.h>
#include <hip/hip_bf16.h>

typedef __attribute__((ext_vector_type(16))) _Float16 v16h;
typedef __attribute__((ext_vector_type(8)))  _Float16 v8h;
typedef __attribute__((ext_vector_type(8)))  float    v8f;
typedef unsigned int uint32;

// ---------------- problem constants ----------------
#define BATCH   4
#define CCH     32
#define HWPIX   65536           // H*W = 256*256
#define NPIX    262144          // B*H*W
#define SCLS    8
#define TILE    128
#define NTILES  2048            // NPIX / TILE
#define K1BLK   512
#define QN      256             // NUM_QUERIES
#define NEGN    256             // NUM_NEGATIVES

// ---------------- workspace layout (bytes) ----------------
static constexpr size_t SZ_PT    = (size_t)NPIX * CCH * 4;      // 33 554 432
static constexpr size_t OFF_P    = 0;
static constexpr size_t OFF_T    = OFF_P + SZ_PT;
static constexpr size_t OFF_TBLV = OFF_T + SZ_PT;               // 8*N int32
static constexpr size_t OFF_TBLH = OFF_TBLV + (size_t)SCLS * NPIX * 4;
static constexpr size_t OFF_PINF = OFF_TBLH + (size_t)SCLS * NPIX * 4; // N u8
static constexpr size_t OFF_TCNT = OFF_PINF + NPIX;             // 2048*8 u32
static constexpr size_t OFF_THCN = OFF_TCNT + (size_t)NTILES * 8 * 4;
static constexpr size_t OFF_PPAR = OFF_THCN + (size_t)NTILES * 8 * 4; // 512*768 f32
static constexpr size_t OFF_PACC = OFF_PPAR + (size_t)K1BLK * 768 * 4; // 768 f32
static constexpr size_t OFF_PRTP = OFF_PACC + 768 * 4;          // 8*32 f32
static constexpr size_t OFF_PRTT = OFF_PRTP + 256 * 4;
static constexpr size_t OFF_CNT  = OFF_PRTT + 256 * 4;          // 8 u32
static constexpr size_t OFF_HCNT = OFF_CNT + 64;
static constexpr size_t OFF_OK   = OFF_HCNT + 64;               // 8 i32
static constexpr size_t OFF_IVN  = OFF_OK + 64;                 // 1 f32 (+pad)
static constexpr size_t OFF_SIML = OFF_IVN + 64;                // 8*7 f32 (+pad)
static constexpr size_t OFF_KEYS = OFF_SIML + 256;              // 8*6 u32
static constexpr size_t OFF_APIX = OFF_KEYS + 256;              // 8*256 i32
static constexpr size_t OFF_CE   = OFF_APIX + (size_t)SCLS * QN * 4; // 2048 f32
static constexpr size_t WS_NEED  = OFF_CE + (size_t)SCLS * QN * 4;

// ---------------- Threefry-2x32 (JAX-compatible schedule) ----------------
__device__ __forceinline__ uint32 rotl32(uint32 v, int n) { return (v << n) | (v >> (32 - n)); }

__device__ __forceinline__ void tf2x32(uint32 k0, uint32 k1, uint32 c0, uint32 c1,
                                       uint32 &r0, uint32 &r1) {
  uint32 ks2 = 0x1BD11BDAu ^ k0 ^ k1;
  uint32 x0 = c0 + k0, x1 = c1 + k1;
#define TFR(rot) { x0 += x1; x1 = rotl32(x1, rot); x1 ^= x0; }
  TFR(13) TFR(15) TFR(26) TFR(6)   x0 += k1;  x1 += ks2 + 1u;
  TFR(17) TFR(29) TFR(16) TFR(24)  x0 += ks2; x1 += k0 + 2u;
  TFR(13) TFR(15) TFR(26) TFR(6)   x0 += k0;  x1 += k1 + 3u;
  TFR(17) TFR(29) TFR(16) TFR(24)  x0 += k1;  x1 += ks2 + 4u;
  TFR(13) TFR(15) TFR(26) TFR(6)   x0 += ks2; x1 += k0 + 5u;
#undef TFR
  r0 = x0; r1 = x1;
}

__device__ __forceinline__ float u01(uint32 bits) {
  return __uint_as_float((bits >> 9) | 0x3f800000u) - 1.0f;
}

// ================= K0: zero the per-(class,q) CE array =================
__global__ void k0_zero(float* ceArr) {
  int i = blockIdx.x * 256 + threadIdx.x;
  if (i < SCLS * QN) ceArr[i] = 0.0f;
}

// ================= K1: stream transform + WMMA prototype sums + repack =================
__global__ void __launch_bounds__(256)
k1_transform(const float* __restrict__ Wt, const float* __restrict__ MU,
             const float* __restrict__ SG, const float* __restrict__ LB,
             const float* __restrict__ MK, const float* __restrict__ PR,
             float* __restrict__ Pg, float* __restrict__ Tg,
             unsigned char* __restrict__ pinfo,
             uint32* __restrict__ tcnt, uint32* __restrict__ thcnt,
             float* __restrict__ protoPart) {
  __shared__ _Float16 q16[3][32][136];   // isw, misw, iw  (f16, padded rows)
  __shared__ float    PT[2][32][129];    // P, T f32 (row pad for conflict-free column reads)
  __shared__ uint32   vflag[TILE], hflag[TILE];

  const int t = threadIdx.x;
  const int lane = t & 31, wave = t >> 5;
  const int qi = (wave < 6) ? (wave >> 1) : 0;
  const int chalf = wave & 1;
  v8f acc = {};                          // per-wave WMMA accumulator (classes x 16 channels)

  for (int tile = blockIdx.x; tile < NTILES; tile += gridDim.x) {
    const int base = tile * TILE;
    // ---- phase 0: classify pixels ----
    if (t < TILE) {
      int p = base + t;
      int b = p >> 16, hw = p & 0xFFFF;
      float m = MK[p];
      int s = 0; float lab = 0.0f;
#pragma unroll
      for (int ss = 0; ss < SCLS; ++ss) {
        float lv = LB[((size_t)(b * SCLS + ss)) * HWPIX + hw];
        if (lv > 0.0f) { s = ss; lab = lv; }
      }
      bool valid = (lab * m) > 0.0f;
      bool hard  = valid && (PR[((size_t)(b * SCLS + s)) * HWPIX + hw] < 0.97f);
      vflag[t] = valid ? (1u << s) : 0u;
      hflag[t] = hard  ? (1u << s) : 0u;
      pinfo[p] = (unsigned char)(s | (valid ? 16 : 0) | (hard ? 32 : 0));
    }
    __syncthreads();
    // ---- per-tile class histograms (threads 0..15) ----
    if (t < 8) {
      uint32 c = 0;
      for (int j = 0; j < TILE; ++j) c += (vflag[j] >> t) & 1u;
      tcnt[tile * 8 + t] = c;
    } else if (t < 16) {
      int s = t - 8; uint32 c = 0;
      for (int j = 0; j < TILE; ++j) c += (hflag[j] >> s) & 1u;
      thcnt[tile * 8 + s] = c;
    }
    // ---- phase 1: streamed transform into LDS (coalesced per channel plane) ----
    for (int it = 0; it < 16; ++it) {
      int idx = it * 256 + t;
      int c = idx >> 7, pix = idx & 127;
      int p = base + pix;
      int b = p >> 16, hw = p & 0xFFFF;
      size_t off = ((size_t)(b * CCH + c)) * HWPIX + hw;
      float wv = Wt[off], sv = SG[off], mv = MU[off];
      float isw = wv / sv;
      q16[0][c][pix] = (_Float16)isw;
      q16[1][c][pix] = (_Float16)(isw * mv);
      q16[2][c][pix] = (_Float16)(1.0f / wv);
      PT[0][c][pix] = mv * wv;
      PT[1][c][pix] = sv * wv;
    }
    __syncthreads();
    // ---- phase 2: waves 0-5 WMMA (read-only LDS) || waves 6-7 repack P/T (read-only LDS) ----
    if (t < 192) {
      const int s = lane & 15;
      const int klo = (lane < 16) ? 0 : 8;
#pragma unroll
      for (int it2 = 0; it2 < 4; ++it2) {
        const int kb = it2 * 32;
        v16h av;
#pragma unroll
        for (int j = 0; j < 16; ++j) {
          int k = j + ((j < 8) ? 0 : 8) + ((lane < 16) ? 0 : 8);
          av[j] = (_Float16)(float)((vflag[kb + k] >> s) & 1u);
        }
        const _Float16* bp = &q16[qi][chalf * 16 + s][kb + klo];
        v8h blo = *(const v8h*)bp;
        v8h bhi = *(const v8h*)(bp + 16);
        v16h bv;
#pragma unroll
        for (int j = 0; j < 8; ++j) { bv[j] = blo[j]; bv[j + 8] = bhi[j]; }
        acc = __builtin_amdgcn_wmma_f32_16x16x32_f16(false, av, false, bv,
                                                     (short)0, acc, false, false);
      }
    } else {
      int l = t - 192;                    // 0..63 -> 2 pixels each
#pragma unroll
      for (int pp = 0; pp < 2; ++pp) {
        int pix = l * 2 + pp;
        size_t gp = (size_t)(base + pix) * CCH;
#pragma unroll
        for (int j = 0; j < 8; ++j) {
          float4 v;
          v.x = PT[0][4 * j + 0][pix]; v.y = PT[0][4 * j + 1][pix];
          v.z = PT[0][4 * j + 2][pix]; v.w = PT[0][4 * j + 3][pix];
          *(float4*)(Pg + gp + 4 * j) = v;
          float4 u;
          u.x = PT[1][4 * j + 0][pix]; u.y = PT[1][4 * j + 1][pix];
          u.z = PT[1][4 * j + 2][pix]; u.w = PT[1][4 * j + 3][pix];
          *(float4*)(Tg + gp + 4 * j) = u;
        }
      }
    }
    __syncthreads();
  }
  // ---- epilogue: write per-block partial prototype sums (no atomics: deterministic) ----
  if (t < 192 && lane < 16) {
#pragma unroll
    for (int v = 0; v < 8; ++v)   // D layout: lanes 0-15 hold classes 0-7 in VGPRs 0-7
      protoPart[(size_t)blockIdx.x * 768 + qi * 256 + v * 32 + chalf * 16 + lane] = acc[v];
  }
}

// ================= K2: exclusive scan of tile histograms =================
__global__ void k2_scan(uint32* tcnt, uint32* thcnt, uint32* cnt, uint32* hcnt) {
  __shared__ uint32 lds[256];
  int t = threadIdx.x;
  for (int task = 0; task < 16; ++task) {
    uint32* p = (task < 8) ? tcnt : thcnt;
    int s = task & 7;
    uint32 v[8]; uint32 sum = 0;
#pragma unroll
    for (int j = 0; j < 8; ++j) { v[j] = p[(t * 8 + j) * 8 + s]; sum += v[j]; }
    lds[t] = sum; __syncthreads();
    for (int off = 1; off < 256; off <<= 1) {
      uint32 add = (t >= off) ? lds[t - off] : 0u;
      __syncthreads();
      lds[t] += add;
      __syncthreads();
    }
    uint32 run = lds[t] - sum;
#pragma unroll
    for (int j = 0; j < 8; ++j) { uint32 tmp = v[j]; p[(t * 8 + j) * 8 + s] = run; run += tmp; }
    if (t == 255) { if (task < 8) cnt[s] = run; else hcnt[s] = run; }
    __syncthreads();
  }
}

// ================= K3: ordered rank-table scatter =================
__global__ void __launch_bounds__(128)
k3_scatter(const unsigned char* __restrict__ pinfo,
           const uint32* __restrict__ tOff, const uint32* __restrict__ thOff,
           int* __restrict__ tblV, int* __restrict__ tblH) {
  __shared__ unsigned char pib[TILE];
  int t = threadIdx.x, tile = blockIdx.x;
  int p = tile * TILE + t;
  unsigned char info = pinfo[p];
  pib[t] = info; __syncthreads();
  int s = info & 7;
  if (info & 16) {
    int r = 0;
    for (int j = 0; j < t; ++j) { unsigned char ij = pib[j]; r += (((ij & 7) == s) && (ij & 16)) ? 1 : 0; }
    tblV[s * NPIX + (int)tOff[tile * 8 + s] + r] = p;
  }
  if (info & 32) {
    int r = 0;
    for (int j = 0; j < t; ++j) { unsigned char ij = pib[j]; r += (((ij & 7) == s) && (ij & 32)) ? 1 : 0; }
    tblH[s * NPIX + (int)thOff[tile * 8 + s] + r] = p;
  }
}

// ================= K4: prototypes, similarity logits, RNG keys, anchor sampling =================
__global__ void k4_proto_rng(const float* __restrict__ protoPart, float* protoAcc,
                             float* protoP, float* protoT,
                             const uint32* __restrict__ cnt, const uint32* __restrict__ hcnt,
                             int* ok, float* invVN, float* simlog, uint32* keys,
                             int* apix, const int* __restrict__ tblH) {
  int t = threadIdx.x;
  // fixed-order reduction of per-block partials (deterministic)
  for (int o = t; o < 768; o += 256) {
    float s = 0.0f;
    for (int b = 0; b < K1BLK; ++b) s += protoPart[(size_t)b * 768 + o];
    protoAcc[o] = s;
  }
  __syncthreads();
  {
    int s = t >> 5, c = t & 31;
    float isw = protoAcc[0 * 256 + s * 32 + c];
    float msw = protoAcc[1 * 256 + s * 32 + c];
    float iw  = protoAcc[2 * 256 + s * 32 + c];
    float psig = 1.0f / isw;
    float pmu  = psig * msw;
    float pw   = 1.0f / iw;
    protoP[s * 32 + c] = pmu * pw;   // proto_mu * proto_weight
    protoT[s * 32 + c] = psig * pw;  // proto_sigma * proto_weight
  }
  if (t == 0) {
    int vn = 0;
    for (int s = 0; s < SCLS; ++s) {
      if (cnt[s] > 0) vn++;
      ok[s] = (cnt[s] > 0 && hcnt[s] > 0) ? 1 : 0;
    }
    *invVN = (vn > 0) ? 1.0f / (float)vn : 0.0f;
    // jax.random.key(42) -> split(8) -> per-class split(3)
    uint32 res[16];
    for (int i = 0; i < 8; ++i) { uint32 o0, o1; tf2x32(0u, 42u, (uint32)i, (uint32)(i + 8), o0, o1); res[i] = o0; res[i + 8] = o1; }
    for (int j = 0; j < 8; ++j) {
      uint32 ck0 = res[2 * j], ck1 = res[2 * j + 1];
      uint32 r6[6];
      for (int i = 0; i < 3; ++i) { uint32 o0, o1; tf2x32(ck0, ck1, (uint32)i, (uint32)(i + 3), o0, o1); r6[i] = o0; r6[i + 3] = o1; }
      for (int e = 0; e < 6; ++e) keys[j * 6 + e] = r6[e];
    }
  }
  __syncthreads();
  // similarity logits (sim / TEMP, -inf where count==0)
  if (t < 56) {
    int i = t / 7, j = t % 7;
    int o = (i + 1 + j) & 7;
    float a = 0.0f;
    for (int c = 0; c < 32; ++c) {
      float d  = protoT[i * 32 + c] + protoT[o * 32 + c];
      float df = protoP[i * 32 + c] - protoP[o * 32 + c];
      a += df * df / d + logf(d);
    }
    float sim = -0.5f * a * (1.0f / 32.0f);
    simlog[i * 7 + j] = (cnt[o] > 0) ? sim * 2.0f : -__builtin_inff();
  }
  // anchor sampling: uniform(k1,(Q,)) over tbl_hard
  for (int a = t; a < SCLS * QN; a += 256) {
    int i = a >> 8, q = a & 255;
    uint32 k10 = keys[i * 6 + 0], k11 = keys[i * 6 + 1];
    uint32 pidx = (q < 128) ? (uint32)q : (uint32)(q - 128);
    uint32 o0, o1; tf2x32(k10, k11, pidx, pidx + 128u, o0, o1);
    float u = u01((q < 128) ? o0 : o1);
    int hc = (int)hcnt[i];
    int ra = 0;
    if (hc > 0) {
      ra = (int)(u * (float)hc);
      int mx = hc - 1; ra = ra < mx ? ra : mx; if (ra < 0) ra = 0;
    }
    apix[a] = (hc > 0) ? tblH[i * NPIX + ra] : 0;
  }
}

// ================= K5: fused negative sampling + MLS logits + logsumexp =================
__global__ void __launch_bounds__(288)
k5_loss(const float* __restrict__ Pg, const float* __restrict__ Tg,
        const float* __restrict__ protoP, const float* __restrict__ protoT,
        const int* __restrict__ tblV, const uint32* __restrict__ cnt,
        const int* __restrict__ ok, const float* __restrict__ simlog,
        const uint32* __restrict__ keys, const int* __restrict__ apix,
        float* __restrict__ ceArr) {
  const int i = blockIdx.x >> 8, q = blockIdx.x & 255;
  if (!ok[i]) return;                       // CE entry stays 0 (zeroed by K0)
  __shared__ float sAP[32], sAT[32];
  __shared__ float sLog[512];
  __shared__ float red[256];
  __shared__ int   sApix;
  const int t = threadIdx.x;
  if (t == 0) sApix = apix[i * QN + q];
  for (int idx = 257 + t; idx < 512; idx += 288) sLog[idx] = -__builtin_inff();
  __syncthreads();
  if (t < 32) {
    size_t ap = (size_t)sApix * CCH + t;
    sAP[t] = Pg[ap]; sAT[t] = Tg[ap];
  }
  __syncthreads();
  if (t < 257) {
    const float4* rp; const float4* rt;
    if (t == 0) {
      rp = (const float4*)(protoP + i * 32);
      rt = (const float4*)(protoT + i * 32);
    } else {
      const int neg = t - 1;
      // categorical over 7 classes via gumbel-argmax (uniform(k2,(Q,NEG,7)))
      uint32 k20 = keys[i * 6 + 2], k21 = keys[i * 6 + 3];
      int f0 = (q * NEGN + neg) * 7;
      float best = -__builtin_inff(); int pick = 0;
#pragma unroll
      for (int j = 0; j < 7; ++j) {
        int f = f0 + j;
        uint32 pidx = (f < 229376) ? (uint32)f : (uint32)(f - 229376);
        uint32 o0, o1; tf2x32(k20, k21, pidx, pidx + 229376u, o0, o1);
        float u = u01((f < 229376) ? o0 : o1);
        u = fmaf(u, 1.0f - 1.17549435e-38f, 1.17549435e-38f);
        float g = -__logf(-__logf(u));
        float v = simlog[i * 7 + j] + g;
        if (v > best) { best = v; pick = j; }
      }
      int ncls = (i + 1 + pick) & 7;
      int c2 = (int)cnt[ncls];
      // rank pick: uniform(k3,(Q,NEG))
      uint32 k30 = keys[i * 6 + 4], k31 = keys[i * 6 + 5];
      int f = q * NEGN + neg;
      uint32 pidx = (f < 32768) ? (uint32)f : (uint32)(f - 32768);
      uint32 o0, o1; tf2x32(k30, k31, pidx, pidx + 32768u, o0, o1);
      float u3 = u01((f < 32768) ? o0 : o1);
      int rn = 0;
      if (c2 > 0) {
        rn = (int)(u3 * (float)c2);
        int mx = c2 - 1; rn = rn < mx ? rn : mx; if (rn < 0) rn = 0;
      }
      int npix = (c2 > 0) ? tblV[ncls * NPIX + rn] : 0;
      rp = (const float4*)(Pg + (size_t)npix * CCH);   // 2x128B contiguous, L2-resident
      rt = (const float4*)(Tg + (size_t)npix * CCH);
    }
    float acc = 0.0f;
#pragma unroll
    for (int j = 0; j < 8; ++j) {
      float4 p4 = rp[j]; float4 t4 = rt[j];
      int cb = 4 * j;
      { float d = sAT[cb + 0] + t4.x; float df = sAP[cb + 0] - p4.x; acc += df * df / d + __logf(d); }
      { float d = sAT[cb + 1] + t4.y; float df = sAP[cb + 1] - p4.y; acc += df * df / d + __logf(d); }
      { float d = sAT[cb + 2] + t4.z; float df = sAP[cb + 2] - p4.z; acc += df * df / d + __logf(d); }
      { float d = sAT[cb + 3] + t4.w; float df = sAP[cb + 3] - p4.w; acc += df * df / d + __logf(d); }
    }
    sLog[t] = -acc * (1.0f / 32.0f);   // mls/TEMP = (-0.5*mean)/0.5
  }
  __syncthreads();
  if (t < 256) red[t] = fmaxf(sLog[t], sLog[t + 256]);
  __syncthreads();
  for (int off = 128; off > 0; off >>= 1) {
    if (t < off) red[t] = fmaxf(red[t], red[t + off]);
    __syncthreads();
  }
  float m = red[0];
  __syncthreads();
  if (t < 256) red[t] = __expf(sLog[t] - m) + __expf(sLog[t + 256] - m);
  __syncthreads();
  for (int off = 128; off > 0; off >>= 1) {
    if (t < off) red[t] += red[t + off];
    __syncthreads();
  }
  if (t == 0) ceArr[i * QN + q] = m + __logf(red[0]) - sLog[0];
}

// ================= K6: deterministic final reduction =================
__global__ void k6_final(const float* __restrict__ ceArr, const float* __restrict__ invVN,
                         float* __restrict__ out) {
  __shared__ float red[256];
  int t = threadIdx.x;
  float s = 0.0f;
#pragma unroll
  for (int j = 0; j < 8; ++j) s += ceArr[t + 256 * j];
  red[t] = s; __syncthreads();
  for (int off = 128; off > 0; off >>= 1) {
    if (t < off) red[t] += red[t + off];
    __syncthreads();
  }
  if (t == 0) out[0] = red[0] * (1.0f / (float)QN) * invVN[0];
}

// ================= launcher =================
extern "C" void kernel_launch(void* const* d_in, const int* in_sizes, int n_in,
                              void* d_out, int out_size, void* d_ws, size_t ws_size,
                              hipStream_t stream) {
  const float* Wt = (const float*)d_in[0];
  const float* MU = (const float*)d_in[1];
  const float* SG = (const float*)d_in[2];
  const float* LB = (const float*)d_in[3];
  const float* MK = (const float*)d_in[4];
  const float* PR = (const float*)d_in[5];
  if (ws_size < WS_NEED) return;

  char* ws = (char*)d_ws;
  float*  Pg    = (float*)(ws + OFF_P);
  float*  Tg    = (float*)(ws + OFF_T);
  int*    tblV  = (int*)(ws + OFF_TBLV);
  int*    tblH  = (int*)(ws + OFF_TBLH);
  unsigned char* pinf = (unsigned char*)(ws + OFF_PINF);
  uint32* tcnt  = (uint32*)(ws + OFF_TCNT);
  uint32* thcn  = (uint32*)(ws + OFF_THCN);
  float*  ppar  = (float*)(ws + OFF_PPAR);
  float*  pacc  = (float*)(ws + OFF_PACC);
  float*  prtP  = (float*)(ws + OFF_PRTP);
  float*  prtT  = (float*)(ws + OFF_PRTT);
  uint32* cnt   = (uint32*)(ws + OFF_CNT);
  uint32* hcnt  = (uint32*)(ws + OFF_HCNT);
  int*    okf   = (int*)(ws + OFF_OK);
  float*  ivn   = (float*)(ws + OFF_IVN);
  float*  siml  = (float*)(ws + OFF_SIML);
  uint32* keys  = (uint32*)(ws + OFF_KEYS);
  int*    apix  = (int*)(ws + OFF_APIX);
  float*  ceArr = (float*)(ws + OFF_CE);

  k0_zero<<<8, 256, 0, stream>>>(ceArr);
  k1_transform<<<K1BLK, 256, 0, stream>>>(Wt, MU, SG, LB, MK, PR,
                                          Pg, Tg, pinf, tcnt, thcn, ppar);
  k2_scan<<<1, 256, 0, stream>>>(tcnt, thcn, cnt, hcnt);
  k3_scatter<<<NTILES, 128, 0, stream>>>(pinf, tcnt, thcn, tblV, tblH);
  k4_proto_rng<<<1, 256, 0, stream>>>(ppar, pacc, prtP, prtT, cnt, hcnt,
                                      okf, ivn, siml, keys, apix, tblH);
  k5_loss<<<SCLS * QN, 288, 0, stream>>>(Pg, Tg, prtP, prtT, tblV, cnt,
                                         okf, siml, keys, apix, ceArr);
  k6_final<<<1, 256, 0, stream>>>(ceArr, ivn, (float*)d_out);
}
